// GCN_59098749993497
// MI455X (gfx1250) — matrix operand order repaired and checked by reading, not compiled, not run on hardware
//
#include <hip/hip_runtime.h>
#include <hip/hip_bf16.h>

// GCN (2-layer GraphConv + mean_nodes) for MI455X / gfx1250, wave32.
// GEMMs run on v_wmma_f32_16x16x32_bf16 (fp32 accum); all per-row norm /
// bias / relu transforms are fused into the LDS staging of the A tile.
// W is staged TRANSPOSED in LDS so both A and B fragments are built from
// contiguous 16-byte runs -> ds_load_b128 (no scalar u16 gathers).
// Bounds checks are hoisted to tile level so the hot path has straight-line
// stores. Edge aggregation uses native f32 global atomics (51 MB target is
// L2-resident). Memory-bound budget ~2.5 GB => ~110us at 23.3 TB/s.

typedef __attribute__((ext_vector_type(16))) __bf16 v16bf;
typedef __attribute__((ext_vector_type(8)))  float  v8f;

#ifndef GCN_N_GRAPHS
#define GCN_N_GRAPHS 100
#endif

// ---------------------------------------------------------------- utilities
__global__ void gcn_fill_zero_u32(unsigned* __restrict__ p, long long n) {
  long long i = (long long)blockIdx.x * blockDim.x + threadIdx.x;
  if (i < n) p[i] = 0u;
}

__global__ void gcn_degree_kernel(const int* __restrict__ src,
                                  const int* __restrict__ dst,
                                  int* __restrict__ degi, int nE, int nN) {
  int e = blockIdx.x * blockDim.x + threadIdx.x;
  if (e < nE) {
    atomicAdd(&degi[src[e]], 1);
    atomicAdd(&degi[nN + dst[e]], 1);
  }
}

__global__ void gcn_norm_kernel(const int* __restrict__ degi,
                                float* __restrict__ sn, float* __restrict__ dn,
                                int nN) {
  int i = blockIdx.x * blockDim.x + threadIdx.x;
  if (i < nN) {
    sn[i] = rsqrtf(fmaxf((float)degi[i], 1.0f));
    dn[i] = rsqrtf(fmaxf((float)degi[nN + i], 1.0f));
  }
}

// ------------------------------------------------------------- WMMA GEMM
// out[M,NCOLS] = transform(A[M,128]) @ W[128,NCOLS]
// MODE 0: a = A[r,k] * rowScale[r]                       (layer-1 input)
// MODE 1: a = relu(A[r,k]*preScale[r] + preBias[k]) * rowScale[r]  (layer-2)
// Block: 256 threads (8 waves). 64 rows x 32 cols of output per block;
// each wave produces one 16x16 f32 tile via 4x v_wmma_f32_16x16x32_bf16.
template <int NCOLS, int MODE>
__global__ __launch_bounds__(256)
void gcn_gemm_wmma(const float* __restrict__ A, const float* __restrict__ W,
                   const float* __restrict__ rowScale,
                   const float* __restrict__ preScale,
                   const float* __restrict__ preBias,
                   float* __restrict__ out, int M) {
  // +8 bf16 pad = 16 B: keeps every row 16B-aligned and staggers banks.
  __shared__ __bf16 sA [64][128 + 8];   // A tile, row-major over K
  __shared__ __bf16 sWt[32][128 + 8];   // W tile, TRANSPOSED: [col][K]

  const int tid     = threadIdx.x;
  const int rowBase = blockIdx.x * 64;
  const int colBase = blockIdx.y * 32;
  const bool fullTile = (rowBase + 64 <= M);   // uniform across the block

  // Stage this block's 32 columns of W (all K=128), fp32 -> bf16,
  // transposed so B-fragment reads are K-contiguous.
#pragma unroll
  for (int i = 0; i < 16; ++i) {
    int idx = i * 256 + tid;            // 4096 elems, coalesced over c
    int k = idx >> 5, c = idx & 31;
    sWt[c][k] = (__bf16)W[k * NCOLS + colBase + c];
  }
  // Stage 64 rows of A with the fused per-row transform, fp32 -> bf16.
  if (fullTile) {
#pragma unroll
    for (int i = 0; i < 32; ++i) {
      int idx = i * 256 + tid;          // 8192 elems, coalesced over k
      int k = idx & 127, r = idx >> 7;
      int row = rowBase + r;
      float v = A[(long long)row * 128 + k];
      if (MODE == 1) v = fmaxf(v * preScale[row] + preBias[k], 0.0f);
      v *= rowScale[row];
      sA[r][k] = (__bf16)v;
    }
  } else {
#pragma unroll
    for (int i = 0; i < 32; ++i) {
      int idx = i * 256 + tid;
      int k = idx & 127, r = idx >> 7;
      int row = rowBase + r;
      float v = 0.0f;
      if (row < M) {
        v = A[(long long)row * 128 + k];
        if (MODE == 1) v = fmaxf(v * preScale[row] + preBias[k], 0.0f);
        v *= rowScale[row];
      }
      sA[r][k] = (__bf16)v;
    }
  }
  __syncthreads();

  const int lane = tid & 31;
  const int wid  = tid >> 5;
  const int wr   = wid & 3;             // wave row tile (0..3)
  const int wc   = wid >> 2;            // wave col tile (0..1)
  const int aRow = wr * 16 + (lane & 15);
  const int bCol = wc * 16 + (lane & 15);
  const int kbA  = (lane < 16) ? 0 : 16;  // A-matrix K half-split (matches B)
  const int khB  = (lane < 16) ? 0 : 16;  // B-matrix K half-split

  v8f acc = {};
#pragma unroll
  for (int k0 = 0; k0 < 128; k0 += 32) {
    v16bf a, b;
    // A lane layout: element e <-> K = kbA + e
    // (one contiguous 32-byte run -> 2x ds_load_b128).
#pragma unroll
    for (int e = 0; e < 16; ++e) {
      int vv = e >> 1;
      int kk = vv << 1;
      a[e] = sA[aRow][k0 + kbA + kk + (e & 1)];
    }
    // B lane layout: element e <-> K = k0 + khB + e; with sWt this is one
    // contiguous 32-byte run -> 2x ds_load_b128.
#pragma unroll
    for (int e = 0; e < 16; ++e) {
      b[e] = sWt[bCol][k0 + khB + e];
    }
    acc = __builtin_amdgcn_wmma_f32_16x16x32_bf16(
        /*neg_a=*/false, a, /*neg_b=*/false, b,
        /*c_mod=*/(short)0, acc, /*reuse_a=*/false, /*reuse_b=*/false);
  }

  // Store D: VGPR r -> M = r (+8 for upper lane half); lane%16 -> N.
  // One base pointer, constant row offsets -> straight-line stores in the
  // (overwhelmingly common) full-tile case.
  const int col = colBase + bCol;
  const int rb  = rowBase + wr * 16 + ((lane < 16) ? 0 : 8);
  float* p = out + (long long)rb * NCOLS + col;
  if (fullTile) {
#pragma unroll
    for (int r = 0; r < 8; ++r) p[(long long)r * NCOLS] = acc[r];
  } else {
#pragma unroll
    for (int r = 0; r < 8; ++r) {
      if (rb + r < M) p[(long long)r * NCOLS] = acc[r];
    }
  }
}

// ------------------------------------------------- edge gather/scatter-add
// One thread per (edge, 4-feature quad): float4 gather from h[src], four
// native f32 atomic adds into agg[dst]. Target array is L2-resident.
__global__ void gcn_scatter_add(const float* __restrict__ h,
                                const int* __restrict__ src,
                                const int* __restrict__ dst,
                                float* __restrict__ agg,
                                int nE, int F, int qShift) {
  long long t = (long long)blockIdx.x * blockDim.x + threadIdx.x;
  long long total = (long long)nE << qShift;
  if (t >= total) return;
  int e = (int)(t >> qShift);
  int j = (int)(t & ((1 << qShift) - 1));
  const float4 v = ((const float4*)(h + (long long)src[e] * F))[j];
  float* p = agg + (long long)dst[e] * F + j * 4;
  unsafeAtomicAdd(p + 0, v.x);
  unsafeAtomicAdd(p + 1, v.y);
  unsafeAtomicAdd(p + 2, v.z);
  unsafeAtomicAdd(p + 3, v.w);
}

// ------------------------------------------------------------ mean pooling
__global__ void gcn_pool_accum(const float* __restrict__ agg,
                               const float* __restrict__ dn,
                               const float* __restrict__ b2,
                               const int* __restrict__ gid,
                               float* __restrict__ gsum,
                               int* __restrict__ gcnt, int nN) {
  long long t = (long long)blockIdx.x * blockDim.x + threadIdx.x;
  if (t >= (long long)nN * 64) return;
  int n = (int)(t >> 6), c = (int)(t & 63);
  int g = gid[n];
  float v = agg[(long long)n * 64 + c] * dn[n] + b2[c];
  unsafeAtomicAdd(&gsum[g * 64 + c], v);
  if (c == 0) atomicAdd(&gcnt[g], 1);
}

__global__ void gcn_pool_final(const float* __restrict__ gsum,
                               const int* __restrict__ gcnt,
                               float* __restrict__ out) {
  int t = blockIdx.x * blockDim.x + threadIdx.x;
  if (t < GCN_N_GRAPHS * 64) {
    out[t] = gsum[t] / fmaxf((float)gcnt[t >> 6], 1.0f);
  }
}

// ----------------------------------------------------------------- launch
extern "C" void kernel_launch(void* const* d_in, const int* in_sizes, int n_in,
                              void* d_out, int out_size, void* d_ws,
                              size_t ws_size, hipStream_t stream) {
  const float* x   = (const float*)d_in[0];
  const float* W1  = (const float*)d_in[1];
  const float* b1  = (const float*)d_in[2];
  const float* W2  = (const float*)d_in[3];
  const float* b2  = (const float*)d_in[4];
  const int*   src = (const int*)d_in[5];
  const int*   dst = (const int*)d_in[6];
  const int*   gid = (const int*)d_in[7];

  const int nN = in_sizes[7];   // graph_ids -> N_NODES
  const int nE = in_sizes[5];   // src       -> N_EDGES

  // Workspace layout (floats): deg(2N int) | sn(N) | dn(N) | h(N*128) |
  // agg(N*128) | gsum(100*64) | gcnt(100 int)  ~= 104 MB total.
  float* ws   = (float*)d_ws;
  int*   degi = (int*)ws;
  float* sn   = ws + 2 * (size_t)nN;
  float* dn   = sn + nN;
  float* h    = dn + nN;
  float* agg  = h + (size_t)nN * 128;
  float* gsum = agg + (size_t)nN * 128;
  int*   gcnt = (int*)(gsum + GCN_N_GRAPHS * 64);

  auto cdiv = [](long long a, long long b) { return (int)((a + b - 1) / b); };

  // Degrees + norms.
  gcn_fill_zero_u32<<<cdiv(2LL * nN, 256), 256, 0, stream>>>((unsigned*)degi,
                                                             2LL * nN);
  gcn_degree_kernel<<<cdiv(nE, 256), 256, 0, stream>>>(src, dst, degi, nE, nN);
  gcn_norm_kernel<<<cdiv(nN, 256), 256, 0, stream>>>(degi, sn, dn, nN);

  // Layer 1: h = (x * sn) @ W1  (WMMA bf16, fp32 accum).
  dim3 g1(cdiv(nN, 64), 128 / 32);
  gcn_gemm_wmma<128, 0><<<g1, 256, 0, stream>>>(x, W1, sn, nullptr, nullptr,
                                                h, nN);
  gcn_fill_zero_u32<<<cdiv((long long)nN * 128, 256), 256, 0, stream>>>(
      (unsigned*)agg, (long long)nN * 128);
  gcn_scatter_add<<<cdiv((long long)nE * 32, 256), 256, 0, stream>>>(
      h, src, dst, agg, nE, 128, 5);

  // Layer 2: h2 = (relu(agg*dn + b1) * sn) @ W2, fused into staging.
  dim3 g2(cdiv(nN, 64), 64 / 32);
  gcn_gemm_wmma<64, 1><<<g2, 256, 0, stream>>>(agg, W2, sn, dn, b1, h, nN);
  gcn_fill_zero_u32<<<cdiv((long long)nN * 64, 256), 256, 0, stream>>>(
      (unsigned*)agg, (long long)nN * 64);
  gcn_scatter_add<<<cdiv((long long)nE * 16, 256), 256, 0, stream>>>(
      h, src, dst, agg, nE, 64, 4);

  // Per-graph mean of (agg2*dn + b2).
  gcn_fill_zero_u32<<<cdiv(GCN_N_GRAPHS * 65, 256), 256, 0, stream>>>(
      (unsigned*)gsum, GCN_N_GRAPHS * 65);
  gcn_pool_accum<<<cdiv((long long)nN * 64, 256), 256, 0, stream>>>(
      agg, dn, b2, gid, gsum, gcnt, nN);
  gcn_pool_final<<<cdiv(GCN_N_GRAPHS * 64, 256), 256, 0, stream>>>(
      gsum, gcnt, (float*)d_out);
}
